// MixingBlock_27762668601562
// MI455X (gfx1250) — matrix operand order repaired and checked by reading, not compiled, not run on hardware
//
#include <hip/hip_runtime.h>

// ---------------------------------------------------------------------------
// MixingBlock on MI455X (gfx1250): all GEMMs via v_wmma_f32_16x16x32_bf16.
// Weights pre-swizzled once into bf16 WMMA-B fragment-major layout; B loads
// are b128 pairs, batched per k-step so loadcnt waits stagger across WMMAs.
// ---------------------------------------------------------------------------

typedef __attribute__((ext_vector_type(16))) __bf16 v16bf;
typedef __attribute__((ext_vector_type(8)))  float  v8f;

#define WMMA_BF16(a, b, c) \
  __builtin_amdgcn_wmma_f32_16x16x32_bf16(false, (a), false, (b), (short)0, (c), false, false)

// A-matrix fragment (16x32 bf16) from fp32 row-major source.
// Lane L: row=L&15; e0..15 -> k0+base+e, base=16*(L>=16)
__device__ __forceinline__ v16bf ldA_f32(const float* __restrict__ src, int lda,
                                         int row0, int k0, int lane) {
  const float* p = src + (size_t)(row0 + (lane & 15)) * lda + k0 + ((lane >> 4) << 4);
  v16bf a;
#pragma unroll
  for (int e = 0; e < 8; ++e) a[e] = (__bf16)p[e];
#pragma unroll
  for (int e = 0; e < 8; ++e) a[8 + e] = (__bf16)p[8 + e];
  return a;
}

// A-matrix fragment from bf16 row-major source (two contiguous 16B loads/lane).
__device__ __forceinline__ v16bf ldA_bf16(const __bf16* __restrict__ src, int lda,
                                          int row0, int k0, int lane) {
  const __bf16* p = src + (size_t)(row0 + (lane & 15)) * lda + k0 + ((lane >> 4) << 4);
  v16bf a;
#pragma unroll
  for (int e = 0; e < 8; ++e) a[e] = p[e];
#pragma unroll
  for (int e = 0; e < 8; ++e) a[8 + e] = p[8 + e];
  return a;
}

// B-matrix fragment from pre-swizzled weights: tile (tk,tn) stored as
// 32 lanes x 16 contiguous bf16 (one 1KB contiguous block per fragment).
__device__ __forceinline__ v16bf ldB_sw(const __bf16* __restrict__ wsw, int ntiles,
                                        int tk, int tn, int lane) {
  const __bf16* p = wsw + (((size_t)tk * ntiles + tn) << 9) + (lane << 4);
  v16bf b;
#pragma unroll
  for (int e = 0; e < 16; ++e) b[e] = p[e];
  return b;
}

// ---------------------------------------------------------------------------
// Kernel 0: one-time weight swizzle fp32(K x Nc) -> bf16 fragment-major.
// ---------------------------------------------------------------------------
__global__ __launch_bounds__(256) void swizzle_w_kernel(
    const float* __restrict__ w, int K, int Nc, __bf16* __restrict__ out) {
  int idx = blockIdx.x * 256 + threadIdx.x;
  if (idx >= K * Nc) return;
  int k = idx / Nc, n = idx - k * Nc;
  int tk = k >> 5, tn = n >> 4;
  int lane = (n & 15) + (k & 16);
  int e = k & 15;
  out[(((size_t)tk * (Nc >> 4) + tn) << 9) + (lane << 4) + e] = (__bf16)w[idx];
}

// ---------------------------------------------------------------------------
// Kernel 1: projection (512->256) + bias + LayerNorm, bf16 output.
// One WG = 16 rows x 256 cols. Wave w owns cols [w*64, w*64+64).
// ---------------------------------------------------------------------------
__global__ __launch_bounds__(128) void proj_ln_kernel(
    const float* __restrict__ x, const float* __restrict__ tmpl,
    const __bf16* __restrict__ wsw_ps, const float* __restrict__ b_ps,
    const float* __restrict__ g_ps, const float* __restrict__ be_ps,
    const __bf16* __restrict__ wsw_pt, const float* __restrict__ b_pt,
    const float* __restrict__ g_pt, const float* __restrict__ be_pt,
    __bf16* __restrict__ xattn) {
  __shared__ float tile[16][256];
  __shared__ float red[16][8][2];

  const int lane = threadIdx.x & 31;
  const int wave = threadIdx.x >> 5;
  const int row0 = blockIdx.x * 16;               // global row tile (template first)
  const bool is_t = row0 < 512 * 64;
  const float* src = is_t ? (tmpl + (size_t)row0 * 512)
                          : (x + (size_t)(row0 - 512 * 64) * 512);
  const __bf16* w = is_t ? wsw_pt : wsw_ps;
  const float* bb = is_t ? b_pt  : b_ps;
  const float* g  = is_t ? g_pt  : g_ps;
  const float* be = is_t ? be_pt : be_ps;

  v8f acc[4] = {};
  for (int kk = 0; kk < 16; ++kk) {               // K = 512 = 16 * 32
    v16bf a  = ldA_f32(src, 512, 0, kk * 32, lane);
    v16bf b0 = ldB_sw(w, 16, kk, wave * 4 + 0, lane);
    v16bf b1 = ldB_sw(w, 16, kk, wave * 4 + 1, lane);
    v16bf b2 = ldB_sw(w, 16, kk, wave * 4 + 2, lane);
    v16bf b3 = ldB_sw(w, 16, kk, wave * 4 + 3, lane);
    acc[0] = WMMA_BF16(a, b0, acc[0]);
    acc[1] = WMMA_BF16(a, b1, acc[1]);
    acc[2] = WMMA_BF16(a, b2, acc[2]);
    acc[3] = WMMA_BF16(a, b3, acc[3]);
  }

  // Stage GEMM + bias into LDS for the cross-wave LayerNorm.
#pragma unroll
  for (int j = 0; j < 4; ++j) {
    int n = wave * 64 + j * 16 + (lane & 15);
    float bv = bb[n];
#pragma unroll
    for (int r = 0; r < 8; ++r) {
      int m = r + ((lane >> 4) << 3);
      tile[m][n] = acc[j][r] + bv;
    }
  }
  __syncthreads();

  // LN: thread t -> row t>>3, segment t&7 (32 elems each).
  {
    int rr = threadIdx.x >> 3, part = threadIdx.x & 7;
    float s = 0.f, s2 = 0.f;
#pragma unroll
    for (int i = 0; i < 32; ++i) {
      float v = tile[rr][part * 32 + i];
      s += v; s2 += v * v;
    }
    red[rr][part][0] = s; red[rr][part][1] = s2;
  }
  __syncthreads();

  const int r = threadIdx.x >> 3;
  const int part = threadIdx.x & 7;
  float s = 0.f, s2 = 0.f;
#pragma unroll
  for (int p = 0; p < 8; ++p) { s += red[r][p][0]; s2 += red[r][p][1]; }
  const float mu  = s * (1.f / 256.f);
  const float var = s2 * (1.f / 256.f) - mu * mu;
  const float inv = rsqrtf(var + 1e-5f);

  __bf16* dst = xattn + (size_t)(row0 + r) * 256 + part * 32;
  const float* tl = &tile[r][part * 32];
  const float* gg = g + part * 32;
  const float* bee = be + part * 32;
#pragma unroll
  for (int i = 0; i < 32; ++i)
    dst[i] = (__bf16)((tl[i] - mu) * inv * gg[i] + bee[i]);
}

// ---------------------------------------------------------------------------
// Kernel 2: QKV (256 -> 768). grid.y = 0/1/2 selects q/k/v column group.
// q,k stored head-major (b,h,n,32); v stored transposed (b,h,32,n) so the
// attention kernel's P*V B-fragments are contiguous.
// ---------------------------------------------------------------------------
__global__ __launch_bounds__(128) void qkv_kernel(
    const __bf16* __restrict__ xattn, const __bf16* __restrict__ wsw_qkv,
    const float* __restrict__ b_qkv, __bf16* __restrict__ qb,
    __bf16* __restrict__ kb, __bf16* __restrict__ vb) {
  const int lane = threadIdx.x & 31;
  const int wave = threadIdx.x >> 5;
  const int row0 = blockIdx.x * 16;
  const int ct = blockIdx.y;                       // which of q/k/v

  v8f acc[4] = {};
  for (int kk = 0; kk < 8; ++kk) {                 // K = 256
    v16bf a  = ldA_bf16(xattn, 256, row0, kk * 32, lane);
    v16bf b0 = ldB_sw(wsw_qkv, 48, kk, ct * 16 + wave * 4 + 0, lane);
    v16bf b1 = ldB_sw(wsw_qkv, 48, kk, ct * 16 + wave * 4 + 1, lane);
    v16bf b2 = ldB_sw(wsw_qkv, 48, kk, ct * 16 + wave * 4 + 2, lane);
    v16bf b3 = ldB_sw(wsw_qkv, 48, kk, ct * 16 + wave * 4 + 3, lane);
    acc[0] = WMMA_BF16(a, b0, acc[0]);
    acc[1] = WMMA_BF16(a, b1, acc[1]);
    acc[2] = WMMA_BF16(a, b2, acc[2]);
    acc[3] = WMMA_BF16(a, b3, acc[3]);
  }

  __bf16* outb = (ct == 0) ? qb : ((ct == 1) ? kb : vb);
#pragma unroll
  for (int j = 0; j < 4; ++j) {
    int cl = wave * 64 + j * 16 + (lane & 15);     // col within this 256 group
    int h = cl >> 5, d = cl & 31;
    float bv = b_qkv[ct * 256 + cl];
#pragma unroll
    for (int r = 0; r < 8; ++r) {
      int grow = row0 + r + ((lane >> 4) << 3);
      int bwin = grow >> 6, nq = grow & 63;
      float val = acc[j][r] + bv;
      if (ct == 2)
        outb[(((size_t)bwin * 8 + h) * 32 + d) * 64 + nq] = (__bf16)val;
      else
        outb[(((size_t)bwin * 8 + h) * 64 + nq) * 32 + d] = (__bf16)val;
    }
  }
}

// ---------------------------------------------------------------------------
// Kernel 3: windowed attention, one WG per (window, head).
// S = q k^T * scale + relpos bias; softmax; O = P v.  head_dim=32 = 1 K-step.
// ---------------------------------------------------------------------------
__global__ __launch_bounds__(128) void attn_kernel(
    const __bf16* __restrict__ qb, const __bf16* __restrict__ kb,
    const __bf16* __restrict__ vb, const float* __restrict__ rpb,
    __bf16* __restrict__ outb) {
  __shared__ __bf16 P[64][64];

  const int lane = threadIdx.x & 31;
  const int wave = threadIdx.x >> 5;
  const int bh = blockIdx.x;
  const int h = bh & 7;
  const int b = bh >> 3;
  const __bf16* q = qb + (size_t)bh * 64 * 32;
  const __bf16* k = kb + (size_t)bh * 64 * 32;
  const __bf16* v = vb + (size_t)bh * 32 * 64;     // transposed (d, n)
  const float scale = 0.17677669529663689f;        // 32^-0.5

  // Queries for this wave's 16 rows as WMMA-A (16x32).
  v16bf aq = ldA_bf16(q, 32, wave * 16, 0, lane);

  // K^T fragments: element e -> kdim = 16*(lane>=16)+e, key = j*16+(lane&15)
  v16bf bk[4];
#pragma unroll
  for (int j = 0; j < 4; ++j) {
    const __bf16* p = k + (size_t)(j * 16 + (lane & 15)) * 32 + ((lane >> 4) << 4);
#pragma unroll
    for (int e = 0; e < 16; ++e) bk[j][e] = p[e];
  }

  v8f sc[4];
#pragma unroll
  for (int j = 0; j < 4; ++j) {
    v8f c = {};
    sc[j] = WMMA_BF16(aq, bk[j], c);
  }

  float s[4][8];
#pragma unroll
  for (int j = 0; j < 4; ++j) {
#pragma unroll
    for (int r = 0; r < 8; ++r) {
      int nq = wave * 16 + r + ((lane >> 4) << 3);
      int nk = j * 16 + (lane & 15);
      int rq = nq >> 3, cq = nq & 7, rk = nk >> 3, ck = nk & 7;
      int idx = (rq - rk + 7) * 15 + (cq - ck + 7);
      s[j][r] = sc[j][r] * scale + rpb[idx * 8 + h];
    }
  }

  // Softmax per query row: 64 keys live in 4 accumulators x 16 lanes of one
  // half-wave (xor masks 1..8 stay inside the half).
#pragma unroll
  for (int r = 0; r < 8; ++r) {
    float m = fmaxf(fmaxf(s[0][r], s[1][r]), fmaxf(s[2][r], s[3][r]));
#pragma unroll
    for (int off = 1; off < 16; off <<= 1) m = fmaxf(m, __shfl_xor(m, off, 32));
    float sum = 0.f;
#pragma unroll
    for (int j = 0; j < 4; ++j) { s[j][r] = __expf(s[j][r] - m); sum += s[j][r]; }
#pragma unroll
    for (int off = 1; off < 16; off <<= 1) sum += __shfl_xor(sum, off, 32);
    float inv = 1.f / sum;
    int nq = wave * 16 + r + ((lane >> 4) << 3);
#pragma unroll
    for (int j = 0; j < 4; ++j)
      P[nq][j * 16 + (lane & 15)] = (__bf16)(s[j][r] * inv);
  }
  __syncthreads();

  // O = P (64x64) x V (64x32): K = 64 = 2 steps, N = 32 = 2 tiles.
  v8f o[2] = {};
#pragma unroll
  for (int kkp = 0; kkp < 2; ++kkp) {
    v16bf ap;
    {
      const __bf16* p = &P[wave * 16 + (lane & 15)][kkp * 32 + ((lane >> 4) << 4)];
#pragma unroll
      for (int e = 0; e < 8; ++e) ap[e] = p[e];
#pragma unroll
      for (int e = 0; e < 8; ++e) ap[8 + e] = p[8 + e];
    }
    // B element (key,d) from transposed v: addr = d*64 + key (contiguous in e)
    v16bf bv0, bv1;
    const __bf16* pv0 = v + (size_t)(0 * 16 + (lane & 15)) * 64
                          + kkp * 32 + ((lane >> 4) << 4);
    const __bf16* pv1 = v + (size_t)(1 * 16 + (lane & 15)) * 64
                          + kkp * 32 + ((lane >> 4) << 4);
#pragma unroll
    for (int e = 0; e < 16; ++e) bv0[e] = pv0[e];
#pragma unroll
    for (int e = 0; e < 16; ++e) bv1[e] = pv1[e];
    o[0] = WMMA_BF16(ap, bv0, o[0]);
    o[1] = WMMA_BF16(ap, bv1, o[1]);
  }

  // Store token-major (b, n, 256) bf16 for the output projection.
#pragma unroll
  for (int jt = 0; jt < 2; ++jt) {
    int d = jt * 16 + (lane & 15);
#pragma unroll
    for (int r = 0; r < 8; ++r) {
      int nq = wave * 16 + r + ((lane >> 4) << 3);
      outb[((size_t)b * 64 + nq) * 256 + h * 32 + d] = (__bf16)o[jt][r];
    }
  }
}

// ---------------------------------------------------------------------------
// Kernel 4: output projection (256 -> 512), per-stream weights, fp32 out.
// Concatenated d_out layout == global row order (template rows first).
// ---------------------------------------------------------------------------
__global__ __launch_bounds__(128) void outproj_kernel(
    const __bf16* __restrict__ attno, const __bf16* __restrict__ wsw_tr,
    const float* __restrict__ b_tr, const __bf16* __restrict__ wsw_sr,
    const float* __restrict__ b_sr, float* __restrict__ out) {
  const int lane = threadIdx.x & 31;
  const int wave = threadIdx.x >> 5;
  const int row0 = blockIdx.x * 16;
  const int ct = blockIdx.y;                       // 0/1 -> cols [0,256)/[256,512)
  const bool is_t = row0 < 512 * 64;
  const __bf16* w = is_t ? wsw_tr : wsw_sr;
  const float* bb = is_t ? b_tr : b_sr;

  v8f acc[4] = {};
  for (int kk = 0; kk < 8; ++kk) {                 // K = 256
    v16bf a  = ldA_bf16(attno, 256, row0, kk * 32, lane);
    v16bf b0 = ldB_sw(w, 32, kk, ct * 16 + wave * 4 + 0, lane);
    v16bf b1 = ldB_sw(w, 32, kk, ct * 16 + wave * 4 + 1, lane);
    v16bf b2 = ldB_sw(w, 32, kk, ct * 16 + wave * 4 + 2, lane);
    v16bf b3 = ldB_sw(w, 32, kk, ct * 16 + wave * 4 + 3, lane);
    acc[0] = WMMA_BF16(a, b0, acc[0]);
    acc[1] = WMMA_BF16(a, b1, acc[1]);
    acc[2] = WMMA_BF16(a, b2, acc[2]);
    acc[3] = WMMA_BF16(a, b3, acc[3]);
  }

#pragma unroll
  for (int j = 0; j < 4; ++j) {
    int c = ct * 256 + wave * 64 + j * 16 + (lane & 15);
    float bv = bb[c];
#pragma unroll
    for (int r = 0; r < 8; ++r) {
      int grow = row0 + r + ((lane >> 4) << 3);
      out[(size_t)grow * 512 + c] = acc[j][r] + bv;
    }
  }
}

// ---------------------------------------------------------------------------
extern "C" void kernel_launch(void* const* d_in, const int* in_sizes, int n_in,
                              void* d_out, int out_size, void* d_ws, size_t ws_size,
                              hipStream_t stream) {
  (void)in_sizes; (void)n_in; (void)out_size; (void)ws_size;

  const float* x     = (const float*)d_in[0];
  const float* tmpl  = (const float*)d_in[1];
  // d_in[2..5] = s_H, s_W, t_H, t_W (unused by the reference computation)
  const float* w_ps  = (const float*)d_in[6];
  const float* b_ps  = (const float*)d_in[7];
  const float* g_ps  = (const float*)d_in[8];
  const float* be_ps = (const float*)d_in[9];
  const float* w_pt  = (const float*)d_in[10];
  const float* b_pt  = (const float*)d_in[11];
  const float* g_pt  = (const float*)d_in[12];
  const float* be_pt = (const float*)d_in[13];
  const float* w_qkv = (const float*)d_in[14];
  const float* b_qkv = (const float*)d_in[15];
  const float* rpb   = (const float*)d_in[16];
  const float* w_tr  = (const float*)d_in[17];
  const float* b_tr  = (const float*)d_in[18];
  const float* w_sr  = (const float*)d_in[19];
  const float* b_sr  = (const float*)d_in[20];

  const size_t ROWS = 2560ull * 64;                // 163840 tokens total
  const size_t CE   = ROWS * 256;                  // elems per 256-wide tensor

  __bf16* xattn  = (__bf16*)d_ws;                  // [ROWS,256] bf16
  __bf16* qb     = xattn + CE;                     // [2560,8,64,32] bf16
  __bf16* kb     = qb + CE;
  __bf16* vb     = kb + CE;                        // [2560,8,32,64] (transposed)
  __bf16* wsw_ps = vb + CE;                        // swizzled bf16 weights
  __bf16* wsw_pt = wsw_ps + 512 * 256;
  __bf16* wsw_qkv= wsw_pt + 512 * 256;
  __bf16* wsw_tr = wsw_qkv + 256 * 768;
  __bf16* wsw_sr = wsw_tr + 256 * 512;
  __bf16* attno  = xattn;                          // reuse: xattn dead after QKV

  // One-time weight swizzles (tiny; weights stay L2-resident afterwards).
  swizzle_w_kernel<<<dim3((512 * 256 + 255) / 256), 256, 0, stream>>>(w_ps, 512, 256, wsw_ps);
  swizzle_w_kernel<<<dim3((512 * 256 + 255) / 256), 256, 0, stream>>>(w_pt, 512, 256, wsw_pt);
  swizzle_w_kernel<<<dim3((256 * 768 + 255) / 256), 256, 0, stream>>>(w_qkv, 256, 768, wsw_qkv);
  swizzle_w_kernel<<<dim3((256 * 512 + 255) / 256), 256, 0, stream>>>(w_tr, 256, 512, wsw_tr);
  swizzle_w_kernel<<<dim3((256 * 512 + 255) / 256), 256, 0, stream>>>(w_sr, 256, 512, wsw_sr);

  proj_ln_kernel<<<dim3((unsigned)(ROWS / 16)), 128, 0, stream>>>(
      x, tmpl, wsw_ps, b_ps, g_ps, be_ps, wsw_pt, b_pt, g_pt, be_pt, xattn);
  qkv_kernel<<<dim3((unsigned)(ROWS / 16), 3), 128, 0, stream>>>(
      xattn, wsw_qkv, b_qkv, qb, kb, vb);
  attn_kernel<<<dim3(2560u * 8u), 128, 0, stream>>>(qb, kb, vb, rpb, attno);
  outproj_kernel<<<dim3((unsigned)(ROWS / 16), 2), 128, 0, stream>>>(
      attno, wsw_tr, b_tr, wsw_sr, b_sr, (float*)d_out);
}
